// _RPN_31679678775605
// MI455X (gfx1250) — matrix operand order repaired and checked by reading, not compiled, not run on hardware
//
#include <hip/hip_runtime.h>
#include <hip/hip_bf16.h>
#include <math.h>
#include <stdint.h>

// ---- CDNA5 wave32 WMMA / TDM types ----
typedef __attribute__((ext_vector_type(16))) _Float16 h16;
typedef __attribute__((ext_vector_type(8)))  _Float16 h8;
typedef __attribute__((ext_vector_type(8)))  float    f8;
typedef unsigned int u32x4 __attribute__((ext_vector_type(4)));
typedef int          i32x4 __attribute__((ext_vector_type(4)));
typedef int          i32x8 __attribute__((ext_vector_type(8)));

#define NEGV  (-1.0e10f)

// ---------------------------------------------------------------------
// TDM: 1-D contiguous tensor load (f16 elements) global -> LDS.
// Builds D# group0/group1 per CDNA5 ISA 8.3/8.4; groups 2/3 unused (2D max).
// ---------------------------------------------------------------------
__device__ __forceinline__
void tdm_load_1d(unsigned lds_addr, const void* gptr, unsigned nelem) {
    unsigned long long ga = (unsigned long long)(uintptr_t)gptr;
    u32x4 g0;
    g0[0] = 1u;                                   // count=1 (valid descriptor)
    g0[1] = lds_addr;                             // lds_addr (bytes)
    g0[2] = (unsigned)ga;                         // global_addr[31:0]
    g0[3] = (unsigned)(ga >> 32) | (2u << 30);    // global_addr[56:32] | type=2
    i32x8 g1;
    g1[0] = (int)(1u << 16);                      // data_size=1 -> 2 bytes (f16)
    g1[1] = (int)((nelem & 0xFFFFu) << 16);       // tensor_dim0[15:0]
    g1[2] = (int)((nelem >> 16) | (1u << 16));    // tensor_dim0[31:16] | tensor_dim1=1
    g1[3] = (int)((nelem & 0xFFFFu) << 16);       // tile_dim0 (<= 65535 elements)
    g1[4] = 1;                                    // tile_dim1=1, tile_dim2=0
    g1[5] = (int)nelem;                           // tensor_dim0_stride[31:0]
    g1[6] = 0;
    g1[7] = 0;
    i32x4 z4 = {0, 0, 0, 0};
#if defined(__clang_major__) && (__clang_major__ >= 23)
    i32x8 z8 = {0, 0, 0, 0, 0, 0, 0, 0};
    __builtin_amdgcn_tensor_load_to_lds(g0, g1, z4, z4, z8, 0);
#else
    __builtin_amdgcn_tensor_load_to_lds(g0, g1, z4, z4, 0);
#endif
}

// =====================================================================
// Prep 0a: weights f32 -> f16, swizzled to the conv LDS tile layout:
// WF16[co_tile 8][ci_chunk 16][tap 9][co 64][ci 32]
// =====================================================================
__global__ __launch_bounds__(256)
void prep_weights(const float* __restrict__ wpre, _Float16* __restrict__ wf) {
    int gid = blockIdx.x * blockDim.x + threadIdx.x;
    if (gid >= 8 * 16 * 9 * 64 * 32) return;
    int ci = gid & 31;
    int t  = gid >> 5;
    int co = t & 63;  t >>= 6;
    int tap = t % 9;  t /= 9;
    int cc = t & 15;
    int ct = t >> 4;
    float v = wpre[((size_t)(ct * 64 + co) * 512 + (size_t)(cc * 32 + ci)) * 9 + tap];
    wf[gid] = (_Float16)v;
}

// =====================================================================
// Prep 0b: input f32 -> f16, swizzled + zero-padded:
// XF16[b 8][ci_chunk 16][Y 66][col 66][ch 32], Y/col index 0..65 <-> y/x -1..64
// A (h, chunk) conv stage is then one contiguous 3*66*32-element block.
// =====================================================================
__global__ __launch_bounds__(256)
void prep_input(const float* __restrict__ xin, _Float16* __restrict__ xf) {
    int gid = blockIdx.x * blockDim.x + threadIdx.x;
    if (gid >= 8 * 16 * 66 * 66 * 32) return;
    int ch  = gid & 31;
    int t   = gid >> 5;
    int col = t % 66; t /= 66;
    int Y   = t % 66; t /= 66;
    int cc  = t & 15;
    int b   = t >> 4;
    int y  = Y - 1;
    int xc = col - 1;
    float v = 0.0f;
    if (y >= 0 && y < 64 && xc >= 0 && xc < 64)
        v = xin[(((size_t)b * 512 + cc * 32 + ch) * 64 + y) * 64 + xc];
    xf[gid] = (_Float16)v;
}

// =====================================================================
// Kernel 1: 3x3 conv (512->512) + bias + ReLU via WMMA f16 implicit GEMM.
// Staging via TDM tensor_load_to_lds (wave 0 issues, s_wait_tensorcnt fence).
// grid: (8 co-tiles, 64 h, 8 b), block: 128 threads = 4 waves (wave32)
// =====================================================================
__global__ __launch_bounds__(128)
void conv3x3_wmma(const _Float16* __restrict__ wf16,
                  const _Float16* __restrict__ xf16,
                  const float* __restrict__ bpre,
                  float* __restrict__ xout) {
    __shared__ _Float16 lds_w[9 * 64 * 32];   // [tap][co 64][ci 32]  36864 B
    __shared__ _Float16 lds_x[3 * 66 * 32];   // [row 3][col 66][ch 32] 12672 B

    const int tid  = threadIdx.x;
    const int lane = tid & 31;           // wave32 (gfx1250)
    const int wv   = tid >> 5;           // wave id 0..3
    const int ct   = blockIdx.x;         // co tile
    const int h    = blockIdx.y;
    const int b    = blockIdx.z;

    f8 acc[4] = {};

    const int m    = lane & 15;
    const int hsel = lane >> 4;

    const unsigned lw_addr = (unsigned)(uintptr_t)&lds_w[0];  // LDS byte offset
    const unsigned lx_addr = (unsigned)(uintptr_t)&lds_x[0];

    for (int cc = 0; cc < 16; ++cc) {
        // ---- TDM DMA stage: one flat copy each for weights and input slab ----
        if (wv == 0) {
            const _Float16* wsrc = wf16 + ((size_t)(ct * 16 + cc)) * (9 * 64 * 32);
            const _Float16* xsrc = xf16 + ((((size_t)b * 16 + cc) * 66 + h) * 66) * 32;
            tdm_load_1d(lw_addr, wsrc, 9 * 64 * 32);   // 18432 f16
            tdm_load_1d(lx_addr, xsrc, 3 * 66 * 32);   // 6336 f16 (rows h-1..h+1 padded)
            __builtin_amdgcn_s_wait_tensorcnt(0);
        }
        __syncthreads();

        for (int tap = 0; tap < 9; ++tap) {
            const int kh = tap / 3, kw = tap - kh * 3;
            // A fragment (16x32 f16): lane<16 -> K {0..7,16..23}; lane>=16 -> {8..15,24..31}
            const int awoff = tap * 2048 + (wv * 16 + m) * 32 + hsel * 8;
            h8 alo = *(const h8*)&lds_w[awoff];
            h8 ahi = *(const h8*)&lds_w[awoff + 16];
            h16 afrag = __builtin_shufflevector(alo, ahi,
                         0, 1, 2, 3, 4, 5, 6, 7, 8, 9, 10, 11, 12, 13, 14, 15);
            // Load all 4 B fragments first so WMMAs aren't gated one-by-one on DScnt
            h16 bfrag[4];
            #pragma unroll
            for (int t = 0; t < 4; ++t) {
                const int n    = t * 16 + m;
                const int boff = kh * 2112 + (n + kw) * 32 + hsel * 8;
                h8 blo = *(const h8*)&lds_x[boff];
                h8 bhi = *(const h8*)&lds_x[boff + 16];
                bfrag[t] = __builtin_shufflevector(blo, bhi,
                            0, 1, 2, 3, 4, 5, 6, 7, 8, 9, 10, 11, 12, 13, 14, 15);
            }
            #pragma unroll
            for (int t = 0; t < 4; ++t)
                acc[t] = __builtin_amdgcn_wmma_f32_16x16x32_f16(
                            false, afrag, false, bfrag[t], (short)0, acc[t], false, false);
        }
        __syncthreads();
    }

    // C/D layout: VGPR r, lanes 0-15 -> (M=r, N=lane); lanes 16-31 -> (M=8+r, N=lane-16)
    const int mbase = hsel * 8;
    #pragma unroll
    for (int r = 0; r < 8; ++r) {
        const int co   = ct * 64 + wv * 16 + mbase + r;
        const float bs = bpre[co];
        #pragma unroll
        for (int t = 0; t < 4; ++t) {
            const int wcol = t * 16 + m;
            float v = acc[t][r] + bs;
            v = fmaxf(v, 0.0f);
            xout[(((size_t)b * 512 + co) * 64 + h) * 64 + wcol] = v;
        }
    }
}

// =====================================================================
// Kernel 2: fused 1x1 convs (512->18 scores, 512->36 deltas)
// =====================================================================
__global__ __launch_bounds__(256)
void conv1x1_heads(const float* __restrict__ x,
                   const float* __restrict__ wcls, const float* __restrict__ bcls,
                   const float* __restrict__ wbox, const float* __restrict__ bbox,
                   float* __restrict__ sc, float* __restrict__ dl) {
    __shared__ float xs[16 * 64];
    __shared__ float wsm[54 * 16];
    const int tid = threadIdx.x;
    const int w   = tid & 63;
    const int og  = tid >> 6;
    const int h   = blockIdx.x;
    const int b   = blockIdx.y;

    float acc[14];
    #pragma unroll
    for (int j = 0; j < 14; ++j) acc[j] = 0.0f;

    for (int ci0 = 0; ci0 < 512; ci0 += 16) {
        for (int i = tid; i < 16 * 64; i += 256) {
            int ci = i >> 6, ww = i & 63;
            xs[i] = x[(((size_t)b * 512 + ci0 + ci) * 64 + h) * 64 + ww];
        }
        for (int i = tid; i < 54 * 16; i += 256) {
            int o = i >> 4, ci = i & 15;
            wsm[i] = (o < 18) ? wcls[(size_t)o * 512 + ci0 + ci]
                              : wbox[(size_t)(o - 18) * 512 + ci0 + ci];
        }
        __syncthreads();
        #pragma unroll
        for (int j = 0; j < 14; ++j) {
            int o = og + 4 * j;
            if (o < 54) {
                float a = 0.0f;
                #pragma unroll
                for (int ci = 0; ci < 16; ++ci)
                    a += wsm[o * 16 + ci] * xs[ci * 64 + w];
                acc[j] += a;
            }
        }
        __syncthreads();
    }
    const int p = h * 64 + w;
    #pragma unroll
    for (int j = 0; j < 14; ++j) {
        int o = og + 4 * j;
        if (o < 18)
            sc[((size_t)b * 18 + o) * 4096 + p] = acc[j] + bcls[o];
        else if (o < 54)
            dl[((size_t)b * 36 + (o - 18)) * 4096 + p] = acc[j] + bbox[o - 18];
    }
}

// =====================================================================
// Kernel 3: anchor gen + sigmoid(fg-bg) + bbox decode + clip + min-size
// =====================================================================
__global__ __launch_bounds__(256)
void decode_anchors(const float* __restrict__ sc, const float* __restrict__ dl,
                    const float* __restrict__ iminfo,
                    float* __restrict__ boxes, float* __restrict__ prob) {
    const int gid = blockIdx.x * blockDim.x + threadIdx.x;
    if (gid >= 8 * 36864) return;
    const int b = gid / 36864;
    const int i = gid - b * 36864;
    const int p = i / 9;
    const int a = i - p * 9;
    const int h = p >> 6;
    const int w = p & 63;

    const float bg = sc[((size_t)b * 18 + a) * 4096 + p];
    const float fg = sc[((size_t)b * 18 + 9 + a) * 4096 + p];
    float pr = 1.0f / (1.0f + expf(-(fg - bg)));

    const int   ri    = a / 3, si = a - ri * 3;
    const float ratio = (ri == 0) ? 0.5f : ((ri == 1) ? 1.0f : 2.0f);
    const float scale = (si == 0) ? 8.0f : ((si == 1) ? 16.0f : 32.0f);
    const float wr = roundf(sqrtf(256.0f / ratio));
    const float hr = roundf(wr * ratio);
    const float Wd = wr * scale, Hd = hr * scale;
    const float sx = (float)w * 16.0f, sy = (float)h * 16.0f;
    const float ax1 = 7.5f - 0.5f * (Wd - 1.0f) + sx;
    const float ay1 = 7.5f - 0.5f * (Hd - 1.0f) + sy;
    const float ax2 = 7.5f + 0.5f * (Wd - 1.0f) + sx;
    const float ay2 = 7.5f + 0.5f * (Hd - 1.0f) + sy;

    const float d0 = dl[((size_t)b * 36 + 4 * a + 0) * 4096 + p];
    const float d1 = dl[((size_t)b * 36 + 4 * a + 1) * 4096 + p];
    const float d2 = dl[((size_t)b * 36 + 4 * a + 2) * 4096 + p];
    const float d3 = dl[((size_t)b * 36 + 4 * a + 3) * 4096 + p];

    const float aw = ax2 - ax1 + 1.0f, ah = ay2 - ay1 + 1.0f;
    const float cx = ax1 + 0.5f * aw,  cy = ay1 + 0.5f * ah;
    const float pcx = d0 * aw + cx, pcy = d1 * ah + cy;
    const float pw  = expf(d2) * aw, ph = expf(d3) * ah;

    const float imH = iminfo[b * 3 + 0];
    const float imW = iminfo[b * 3 + 1];
    const float scl = iminfo[b * 3 + 2];

    float bx1 = fminf(fmaxf(pcx - 0.5f * pw, 0.0f), imW - 1.0f);
    float by1 = fminf(fmaxf(pcy - 0.5f * ph, 0.0f), imH - 1.0f);
    float bx2 = fminf(fmaxf(pcx + 0.5f * pw, 0.0f), imW - 1.0f);
    float by2 = fminf(fmaxf(pcy + 0.5f * ph, 0.0f), imH - 1.0f);

    const float ws_ = bx2 - bx1 + 1.0f, hs_ = by2 - by1 + 1.0f;
    const float ms  = 16.0f * scl;
    if (!(ws_ >= ms && hs_ >= ms)) pr = NEGV;

    float* bo = boxes + ((size_t)b * 36864 + i) * 4;
    bo[0] = bx1; bo[1] = by1; bo[2] = bx2; bo[3] = by2;
    prob[(size_t)b * 36864 + i] = pr;
}

// =====================================================================
// Kernel 4: greedy NMS -> 300 rois/batch.
// =====================================================================
__global__ __launch_bounds__(1024)
void nms_rois(const float* __restrict__ boxes, float* __restrict__ prob,
              float* __restrict__ out) {
    __shared__ float sval[1024];
    __shared__ int   sidx[1024];
    __shared__ float sbox[4];

    const int b   = blockIdx.x;
    const int tid = threadIdx.x;
    const float* bb = boxes + (size_t)b * 36864 * 4;
    float*       pp = prob  + (size_t)b * 36864;

    for (int it = 0; it < 300; ++it) {
        float best = -1.0e30f; int bj = 0;
        #pragma unroll
        for (int k = 0; k < 36; ++k) {
            int j = tid + k * 1024;
            float v = pp[j];
            if (v > best) { best = v; bj = j; }
        }
        sval[tid] = best; sidx[tid] = bj;
        __syncthreads();
        for (int s = 512; s > 0; s >>= 1) {
            if (tid < s && sval[tid + s] > sval[tid]) {
                sval[tid] = sval[tid + s];
                sidx[tid] = sidx[tid + s];
            }
            __syncthreads();
        }
        if (tid == 0) {
            const int j = sidx[0];
            const bool valid = sval[0] > 0.5f * NEGV;
            const float b0 = bb[j * 4 + 0], b1 = bb[j * 4 + 1];
            const float b2 = bb[j * 4 + 2], b3 = bb[j * 4 + 3];
            sbox[0] = b0; sbox[1] = b1; sbox[2] = b2; sbox[3] = b3;
            float* o = out + ((size_t)b * 300 + it) * 5;
            o[0] = (float)b;
            o[1] = valid ? b0 : 0.0f;
            o[2] = valid ? b1 : 0.0f;
            o[3] = valid ? b2 : 0.0f;
            o[4] = valid ? b3 : 0.0f;
        }
        __syncthreads();
        const float cx1 = sbox[0], cy1 = sbox[1], cx2 = sbox[2], cy2 = sbox[3];
        const float car = (cx2 - cx1 + 1.0f) * (cy2 - cy1 + 1.0f);
        #pragma unroll
        for (int k = 0; k < 36; ++k) {
            int j = tid + k * 1024;
            const float x1 = bb[j * 4 + 0], y1 = bb[j * 4 + 1];
            const float x2 = bb[j * 4 + 2], y2 = bb[j * 4 + 3];
            const float ar = (x2 - x1 + 1.0f) * (y2 - y1 + 1.0f);
            const float iw = fmaxf(fminf(cx2, x2) - fmaxf(cx1, x1) + 1.0f, 0.0f);
            const float ih = fmaxf(fminf(cy2, y2) - fmaxf(cy1, y1) + 1.0f, 0.0f);
            const float inter = iw * ih;
            const float iou = inter / (car + ar - inter);
            if (iou > 0.7f) pp[j] = NEGV;
        }
        __syncthreads();
    }
}

// =====================================================================
extern "C" void kernel_launch(void* const* d_in, const int* in_sizes, int n_in,
                              void* d_out, int out_size, void* d_ws, size_t ws_size,
                              hipStream_t stream) {
    const float* base_feat = (const float*)d_in[0];
    const float* im_info   = (const float*)d_in[1];
    // d_in[2] = gt_boxes (unused by reference output)
    const float* w_pre = (const float*)d_in[3];
    const float* b_pre = (const float*)d_in[4];
    const float* w_cls = (const float*)d_in[5];
    const float* b_cls = (const float*)d_in[6];
    const float* w_box = (const float*)d_in[7];
    const float* b_box = (const float*)d_in[8];
    float* out = (float*)d_out;

    // workspace layout (float units)
    float* X   = (float*)d_ws;                 // 8*512*64*64   = 16,777,216
    float* SC  = X  + (size_t)16777216;        // 8*18*4096     =    589,824
    float* DL  = SC + (size_t)589824;          // 8*36*4096     =  1,179,648
    float* BX  = DL + (size_t)1179648;         // 8*36864*4     =  1,179,648
    float* PB  = BX + (size_t)1179648;         // 8*36864       =    294,912
    _Float16* WF16 = (_Float16*)(PB + (size_t)294912);      // 2,359,296 f16
    _Float16* XF16 = WF16 + (size_t)2359296;                // 17,842,176 f16

    prep_weights<<<(8 * 16 * 9 * 64 * 32 + 255) / 256, 256, 0, stream>>>(w_pre, WF16);
    prep_input<<<(8 * 16 * 66 * 66 * 32 + 255) / 256, 256, 0, stream>>>(base_feat, XF16);
    conv3x3_wmma<<<dim3(8, 64, 8), 128, 0, stream>>>(WF16, XF16, b_pre, X);
    conv1x1_heads<<<dim3(64, 8), 256, 0, stream>>>(X, w_cls, b_cls, w_box, b_box, SC, DL);
    decode_anchors<<<(8 * 36864 + 255) / 256, 256, 0, stream>>>(SC, DL, im_info, BX, PB);
    nms_rois<<<8, 1024, 0, stream>>>(BX, PB, out);
}
